// Narx_23235773071509
// MI455X (gfx1250) — matrix operand order repaired and checked by reading, not compiled, not run on hardware
//
#include <hip/hip_runtime.h>
#include <hip/hip_bf16.h>

typedef __attribute__((ext_vector_type(16))) _Float16 v16h;
typedef __attribute__((ext_vector_type(8)))  _Float16 v8h;
typedef __attribute__((ext_vector_type(8)))  float    v8f;

#define NXC     16
#define HIDC    128
#define NTC     1024
#define NGRIDC  2048
#define TILE_G  128
#define MAXDLY  8
#define TCHUNK  8      // timesteps per workgroup; 1016 = 8 * 127

// Hardware transcendental tanh (CDNA5 TRANS32 op) with safe fallbacks.
static __device__ __forceinline__ float fast_tanh(float x) {
#if __has_builtin(__builtin_amdgcn_tanhf)
  return __builtin_amdgcn_tanhf(x);
#elif __has_builtin(__builtin_amdgcn_tanh_f32)
  return __builtin_amdgcn_tanh_f32(x);
#else
  float r;
  // TRANS ops need 1 independent op / V_NOP before the result is consumed;
  // raw asm bypasses compiler hazard handling, so pad explicitly.
  asm volatile("v_tanh_f32 %0, %1\n\tv_nop\n\tv_nop" : "=v"(r) : "v"(x));
  return r;
#endif
}

// ---------------- workspace layout (bytes) ----------------
// 0      : W_eff  f16 [128][128]   (folded W_in, K = dt*16+ch)
// 32768  : W_ih   f16 [128][128]
// 65536  : b1     f32 [128]
// 66048  : b2     f32 [128]        (b_ih + b_hh)
// 66560  : Wout   f32 [128]
// 67072  : bout   f32 [1]

// Fold the 144-wide delayed-feature weight into a dense 128-wide weight over
// the contiguous window x[t-8..t-1, g, 0:16].  fx offsets = [-1,-8..-1],
// fy offsets = [-1,-8..-1]; offset -1 (dt==7) appears twice.
__global__ void narx_prep(const float* __restrict__ W_in, const float* __restrict__ b_in,
                          const float* __restrict__ W_ih, const float* __restrict__ b_ih,
                          const float* __restrict__ b_hh, const float* __restrict__ W_out,
                          const float* __restrict__ b_out,
                          _Float16* __restrict__ Weff, _Float16* __restrict__ Wih,
                          float* __restrict__ B1, float* __restrict__ B2,
                          float* __restrict__ Wo, float* __restrict__ Bo) {
  const int o = threadIdx.x;
  if (o >= HIDC) return;
  const float* wr = W_in + o * 144;
#pragma unroll
  for (int dt = 0; dt < 8; ++dt) {
    for (int ch = 0; ch < 16; ++ch) {
      float v;
      if (ch < 15) {                       // fx block: feat idx j*15+ch, j=dt+1 (+ j=0 for dt==7)
        v = wr[(dt + 1) * 15 + ch];
        if (dt == 7) v += wr[ch];
      } else {                             // fy block: feat idx 135+j
        v = wr[135 + dt + 1];
        if (dt == 7) v += wr[135];
      }
      Weff[o * HIDC + dt * 16 + ch] = (_Float16)v;
    }
  }
  for (int k = 0; k < HIDC; ++k) Wih[o * HIDC + k] = (_Float16)W_ih[o * HIDC + k];
  B1[o] = b_in[o];
  B2[o] = b_ih[o] + b_hh[o];
  Wo[o] = W_out[o];
  if (o == 0) Bo[0] = b_out[0];
}

// First MAXDLY timesteps of the output are a pass-through of channel 15.
__global__ void narx_head(const float* __restrict__ x, float* __restrict__ out) {
  int i = blockIdx.x * 256 + threadIdx.x;
  if (i < MAXDLY * NGRIDC) {
    int t = i >> 11, g = i & (NGRIDC - 1);
    out[(size_t)t * NGRIDC + g] = x[((size_t)t * NGRIDC + g) * NXC + 15];
  }
}

// A-fragment: lane = lo + 16*hi holds M=lo; elems 0..7 -> K = kb..kb+7,
// elems 8..15 -> K = kb+16..kb+23  (kb = kc*32 + 8*hi).
static __device__ __forceinline__ v16h ldA(const _Float16* p) {
  v8h a = *(const v8h*)p;
  v8h b = *(const v8h*)(p + 16);
  v16h r;
#pragma unroll
  for (int i = 0; i < 8; ++i) { r[i] = a[i]; r[i + 8] = b[i]; }
  return r;
}

__global__ __launch_bounds__(256) void narx_main(
    const float* __restrict__ x,
    const _Float16* __restrict__ Weff, const _Float16* __restrict__ Wih,
    const float* __restrict__ B1, const float* __restrict__ B2,
    const float* __restrict__ Wo, const float* __restrict__ Bo,
    float* __restrict__ out) {
  extern __shared__ __align__(16) char smem[];
  _Float16* sW1 = (_Float16*)smem;                 // 32 KB
  _Float16* sW2 = sW1 + HIDC * HIDC;               // 32 KB
  _Float16* sXw = sW2 + HIDC * HIDC;               // 32 KB circular window [row][slot*16+ch]
  _Float16* sH  = sXw + HIDC * HIDC;               // 32 KB h_in / h (f16)
  float* sB1 = (float*)(sH + HIDC * HIDC);
  float* sB2 = sB1 + HIDC;
  float* sWo = sB2 + HIDC;

  const int t0  = MAXDLY + blockIdx.y * TCHUNK;    // t0 % 8 == 0  ->  phase(t0+i) == i
  const int g0  = blockIdx.x * TILE_G;
  const int tid = threadIdx.x;

  // ---- stage weights + biases into LDS (128-bit copies), once per chunk ----
  {
    const uint4* s1 = (const uint4*)Weff;
    const uint4* s2 = (const uint4*)Wih;
    uint4* d1 = (uint4*)sW1;
    uint4* d2 = (uint4*)sW2;
    for (int i = tid; i < HIDC * HIDC / 8; i += 256) { d1[i] = s1[i]; d2[i] = s2[i]; }
    if (tid < HIDC) { sB1[tid] = B1[tid]; sB2[tid] = B2[tid]; sWo[tid] = Wo[tid]; }
  }

  // ---- preload the 7 oldest window slices: t' = t0-8 .. t0-2, slot = t' & 7 = sl ----
  for (int s = tid; s < 7 * TILE_G; s += 256) {
    const int sl = s >> 7, row = s & (TILE_G - 1);
    const float* src = x + ((size_t)(t0 - MAXDLY + sl) * NGRIDC + (g0 + row)) * NXC;
    _Float16* dst = sXw + row * HIDC + sl * 16;
#pragma unroll
    for (int j = 0; j < 4; ++j) {
      float4 v = *(const float4*)(src + 4 * j);
      dst[4 * j + 0] = (_Float16)v.x; dst[4 * j + 1] = (_Float16)v.y;
      dst[4 * j + 2] = (_Float16)v.z; dst[4 * j + 3] = (_Float16)v.w;
    }
  }

  const int wv = tid >> 5, lane = tid & 31;
  const int lo = lane & 15, hi = lane >> 4;
  const int rowA = wv * 16 + lo;

  v8f zf;
#pragma unroll
  for (int i = 0; i < 8; ++i) zf[i] = 0.0f;

#pragma unroll
  for (int i = 0; i < TCHUNK; ++i) {
    const int t = t0 + i;            // phase = t & 7 = i (compile-time after unroll)

    // ---- slide window: load slice t-1 into slot (i+7)&7 (held t-9, read last iter) ----
    {
      const int slot = (i + 7) & 7;
      for (int s = tid; s < 2 * TILE_G; s += 256) {     // 128 rows x 16 floats, 2 threads/row
        const int row = s >> 1, q = (s & 1) * 2;
        const float* src = x + ((size_t)(t - 1) * NGRIDC + (g0 + row)) * NXC + 4 * q;
        __builtin_prefetch(src + (size_t)NGRIDC * NXC, 0, 1);   // next timestep's slice
        _Float16* dst = sXw + row * HIDC + slot * 16 + 4 * q;
#pragma unroll
        for (int j = 0; j < 2; ++j) {
          float4 v = *(const float4*)(src + 4 * j);
          dst[4 * j + 0] = (_Float16)v.x; dst[4 * j + 1] = (_Float16)v.y;
          dst[4 * j + 2] = (_Float16)v.z; dst[4 * j + 3] = (_Float16)v.w;
        }
      }
    }
    __syncthreads();

    v8f acc[8];
    // ================= layer 1: window(128) x Weff^T (phase-rotated K) =================
#pragma unroll
    for (int n = 0; n < 8; ++n) acc[n] = zf;
#pragma unroll
    for (int kc = 0; kc < 4; ++kc) {
      const v16h a = *(const v16h*)(&sXw[rowA * HIDC + kc * 32 + hi * 16]);
      // physical 16-wide K-chunk s = 2*kc+hi holds timestep with dt = (s - phase) & 7
      const int krot = (((2 * kc + hi) + 8 - i) & 7) * 16;
#pragma unroll
      for (int n = 0; n < 8; ++n) {
        const v16h b = *(const v16h*)(&sW1[(n * 16 + lo) * HIDC + krot]);
        acc[n] = __builtin_amdgcn_wmma_f32_16x16x32_f16(false, a, false, b,
                                                        (short)0, acc[n], false, false);
      }
    }
    // bias + ReLU -> f16 h_in into this wave's own rows of sH
#pragma unroll
    for (int n = 0; n < 8; ++n) {
      const int col = n * 16 + lo;
      const float bn = sB1[col];
#pragma unroll
      for (int e = 0; e < 8; ++e) {
        float v = acc[n][e] + bn;
        v = v > 0.0f ? v : 0.0f;
        sH[(wv * 16 + e + 8 * hi) * HIDC + col] = (_Float16)v;
      }
    }
    __syncthreads();

    // ================= layer 2: h_in(128) x Wih^T =================
#pragma unroll
    for (int n = 0; n < 8; ++n) acc[n] = zf;
#pragma unroll
    for (int kc = 0; kc < 4; ++kc) {
      const v16h a = *(const v16h*)(&sH[rowA * HIDC + kc * 32 + hi * 16]);
#pragma unroll
      for (int n = 0; n < 8; ++n) {
        const v16h b = *(const v16h*)(&sW2[(n * 16 + lo) * HIDC + kc * 32 + hi * 16]);
        acc[n] = __builtin_amdgcn_wmma_f32_16x16x32_f16(false, a, false, b,
                                                        (short)0, acc[n], false, false);
      }
    }
    // bias + hardware tanh -> f16 h into sH (own rows; reads above were own rows too)
#pragma unroll
    for (int n = 0; n < 8; ++n) {
      const int col = n * 16 + lo;
      const float bn = sB2[col];
#pragma unroll
      for (int e = 0; e < 8; ++e) {
        const float v = fast_tanh(acc[n][e] + bn);
        sH[(wv * 16 + e + 8 * hi) * HIDC + col] = (_Float16)v;
      }
    }
    __syncthreads();

    // ================= output head: dot with W_out =================
    if (tid < TILE_G) {
      float s = Bo[0];
      const _Float16* hr = &sH[tid * HIDC];
#pragma unroll
      for (int c = 0; c < HIDC; ++c) s += (float)hr[c] * sWo[c];
      out[(size_t)t * NGRIDC + g0 + tid] = s;
    }
    // next iteration's __syncthreads() (after slice store) orders the head's sH reads
    // of this iteration before the next layer-1 sH writes; the syncs after layer 1 order window reads.
  }
}

extern "C" void kernel_launch(void* const* d_in, const int* in_sizes, int n_in,
                              void* d_out, int out_size, void* d_ws, size_t ws_size,
                              hipStream_t stream) {
  const float* x     = (const float*)d_in[0];
  const float* W_in  = (const float*)d_in[1];
  const float* b_in  = (const float*)d_in[2];
  const float* W_ih  = (const float*)d_in[3];
  const float* b_ih  = (const float*)d_in[4];
  // d_in[5] = W_hh: dead in the reference (only b_hh is used)
  const float* b_hh  = (const float*)d_in[6];
  const float* W_out = (const float*)d_in[7];
  const float* b_out = (const float*)d_in[8];
  float* out = (float*)d_out;

  char* ws = (char*)d_ws;
  _Float16* Weff = (_Float16*)(ws + 0);
  _Float16* Wih  = (_Float16*)(ws + 32768);
  float* B1 = (float*)(ws + 65536);
  float* B2 = (float*)(ws + 66048);
  float* Wo = (float*)(ws + 66560);
  float* Bo = (float*)(ws + 67072);

  narx_prep<<<1, 128, 0, stream>>>(W_in, b_in, W_ih, b_ih, b_hh, W_out, b_out,
                                   Weff, Wih, B1, B2, Wo, Bo);
  narx_head<<<(MAXDLY * NGRIDC + 255) / 256, 256, 0, stream>>>(x, out);

  dim3 grid(NGRIDC / TILE_G, (NTC - MAXDLY) / TCHUNK);   // 16 x 127
  const size_t shmem = 4u * HIDC * HIDC * sizeof(_Float16) + 3u * HIDC * sizeof(float);
  narx_main<<<grid, 256, shmem, stream>>>(x, Weff, Wih, B1, B2, Wo, Bo, out);
}
